// CDSMVSNet_22428319219719
// MI455X (gfx1250) — matrix operand order repaired and checked by reading, not compiled, not run on hardware
//
#include <hip/hip_runtime.h>

typedef __attribute__((ext_vector_type(16))) _Float16 v16h;
typedef __attribute__((ext_vector_type(8)))  float    v8f;

#define Hh 128
#define Ww 160
#define HW (Hh*Ww)          // 20480
#define Cc 32
#define Dd 48
#define NVIEW 3             // num src views

// ---------------------------------------------------------------------------
// K0: projection setup. ref_proj / src_proj are [[A,b],[0,0,0,1]] with
// A = K[:3,:3] @ E[:3,:3], b = K[:3,:3] @ E[:3,3].  P = src @ inv(ref):
// rot = As*inv(Ar), trans = bs - As*inv(Ar)*br.  One thread does all views.
// ---------------------------------------------------------------------------
__device__ void proj_top(const float* p, float A[9], float b3[3]) {
  const float* E = p;        // extrinsic 4x4
  const float* K = p + 16;   // intrinsic 4x4
  for (int i = 0; i < 3; i++)
    for (int j = 0; j < 4; j++) {
      float s = 0.f;
      for (int k = 0; k < 3; k++) s += K[i*4+k] * E[k*4+j];
      if (j < 3) A[i*3+j] = s; else b3[i] = s;
    }
}

__global__ void proj_kernel(const float* __restrict__ proj, float* __restrict__ out) {
  if (threadIdx.x != 0 || blockIdx.x != 0) return;
  float Ar[9], br[3];
  proj_top(proj, Ar, br);
  float det = Ar[0]*(Ar[4]*Ar[8]-Ar[5]*Ar[7])
            - Ar[1]*(Ar[3]*Ar[8]-Ar[5]*Ar[6])
            + Ar[2]*(Ar[3]*Ar[7]-Ar[4]*Ar[6]);
  float id = 1.f / det;
  float inv[9];
  inv[0]=(Ar[4]*Ar[8]-Ar[5]*Ar[7])*id; inv[1]=(Ar[2]*Ar[7]-Ar[1]*Ar[8])*id; inv[2]=(Ar[1]*Ar[5]-Ar[2]*Ar[4])*id;
  inv[3]=(Ar[5]*Ar[6]-Ar[3]*Ar[8])*id; inv[4]=(Ar[0]*Ar[8]-Ar[2]*Ar[6])*id; inv[5]=(Ar[2]*Ar[3]-Ar[0]*Ar[5])*id;
  inv[6]=(Ar[3]*Ar[7]-Ar[4]*Ar[6])*id; inv[7]=(Ar[1]*Ar[6]-Ar[0]*Ar[7])*id; inv[8]=(Ar[0]*Ar[4]-Ar[1]*Ar[3])*id;
  float ib[3];
  for (int i = 0; i < 3; i++) ib[i] = inv[i*3]*br[0] + inv[i*3+1]*br[1] + inv[i*3+2]*br[2];
  for (int v = 0; v < NVIEW; v++) {
    float As[9], bs[3];
    proj_top(proj + (v+1)*32, As, bs);
    float R[9], t3[3];
    for (int i = 0; i < 3; i++)
      for (int j = 0; j < 3; j++) {
        float s = 0.f;
        for (int k = 0; k < 3; k++) s += As[i*3+k]*inv[k*3+j];
        R[i*3+j] = s;
      }
    for (int i = 0; i < 3; i++) {
      float s = 0.f;
      for (int k = 0; k < 3; k++) s += As[i*3+k]*ib[k];
      t3[i] = bs[i] - s;
    }
    for (int i = 0; i < 9; i++) out[v*16+i] = R[i];
    for (int i = 0; i < 3; i++) out[v*16+9+i] = t3[i];
  }
}

// ---------------------------------------------------------------------------
// K1: CHW -> HWC transpose of ref (img 0) and 3 src feature maps (imgs 1..3)
// so a bilinear tap becomes one contiguous 128B read.
// ---------------------------------------------------------------------------
__global__ void chw2hwc_kernel(const float* __restrict__ ref, const float* __restrict__ src,
                               float* __restrict__ out) {
  __shared__ float tile[32][65];
  int img = blockIdx.y;
  int n0  = blockIdx.x * 64;
  const float* in = (img == 0) ? ref : (src + (size_t)(img-1)*Cc*HW);
  float* o = out + (size_t)img*HW*Cc;
  int tid = threadIdx.x;
  for (int it = 0; it < 8; it++) {
    int idx = it*256 + tid;
    int c = idx >> 6, p = idx & 63;
    tile[c][p] = in[(size_t)c*HW + n0 + p];
  }
  __syncthreads();
  for (int it = 0; it < 8; it++) {
    int idx = it*256 + tid;
    int p = idx >> 5, c = idx & 31;
    o[(size_t)(n0+p)*Cc + c] = tile[c][p];
  }
}

// ---------------------------------------------------------------------------
// K2: homography warp + dual channel-dot. One thread = one pixel, 6 depths.
// sim[v][d][n]   = sum_c ref[c,n]*warped[c,d,n]
// cpart[v][d][n] = sum_c reg_w[c]*ref[c,n]*warped[c,d,n]
// ---------------------------------------------------------------------------
__global__ void warp_kernel(const float* __restrict__ projw, const float* __restrict__ hwc,
                            const float* __restrict__ regw, const float* __restrict__ dvals,
                            float* __restrict__ sim, float* __restrict__ cpart) {
  int n  = blockIdx.x * 256 + threadIdx.x;
  int v  = blockIdx.z;
  int d0 = blockIdx.y * 6;
  const float* P = projw + v*16;
  float xf = (float)(n % Ww), yf = (float)(n / Ww);
  float rx = P[0]*xf + P[1]*yf + P[2];
  float ry = P[3]*xf + P[4]*yf + P[5];
  float rz = P[6]*xf + P[7]*yf + P[8];
  float t0 = P[9], t1 = P[10], t2 = P[11];

  const float4* refv = (const float4*)(hwc + (size_t)n*Cc);
  const float*  srcb = hwc + (size_t)(v+1)*HW*Cc;
  float4 rc[8], cw[8];
#pragma unroll
  for (int i = 0; i < 8; i++) {
    rc[i] = refv[i];
    float4 rw = ((const float4*)regw)[i];
    cw[i].x = rc[i].x*rw.x; cw[i].y = rc[i].y*rw.y;
    cw[i].z = rc[i].z*rw.z; cw[i].w = rc[i].w*rw.w;
  }

  for (int dd = 0; dd < 6; dd++) {
    int d = d0 + dd;
    float dep = dvals[(size_t)d*HW + n];
    float px = rx*dep + t0, py = ry*dep + t1, pz = rz*dep + t2;
    float iz = 1.f / pz;
    float xn = px*iz*(2.f/(float)(Ww-1)) - 1.f;
    float yn = py*iz*(2.f/(float)(Hh-1)) - 1.f;
    float xp = ((xn+1.f)*(float)Ww - 1.f)*0.5f;
    float yp = ((yn+1.f)*(float)Hh - 1.f)*0.5f;
    float x0 = floorf(xp), y0 = floorf(yp);
    float fx = xp - x0,   fy = yp - y0;
    float x1f = x0 + 1.f, y1f = y0 + 1.f;
    float vx0 = (x0  >= 0.f && x0  <= (float)(Ww-1)) ? 1.f : 0.f;
    float vx1 = (x1f >= 0.f && x1f <= (float)(Ww-1)) ? 1.f : 0.f;
    float vy0 = (y0  >= 0.f && y0  <= (float)(Hh-1)) ? 1.f : 0.f;
    float vy1 = (y1f >= 0.f && y1f <= (float)(Hh-1)) ? 1.f : 0.f;
    float w00 = (1.f-fx)*(1.f-fy)*vx0*vy0;
    float w10 = fx*(1.f-fy)*vx1*vy0;
    float w01 = (1.f-fx)*fy*vx0*vy1;
    float w11 = fx*fy*vx1*vy1;
    int xc0 = (int)fminf(fmaxf(x0, 0.f), (float)(Ww-1));
    int xc1 = (int)fminf(fmaxf(x1f,0.f), (float)(Ww-1));
    int yc0 = (int)fminf(fmaxf(y0, 0.f), (float)(Hh-1));
    int yc1 = (int)fminf(fmaxf(y1f,0.f), (float)(Hh-1));
    const float4* p00 = (const float4*)(srcb + (size_t)(yc0*Ww + xc0)*Cc);
    const float4* p10 = (const float4*)(srcb + (size_t)(yc0*Ww + xc1)*Cc);
    const float4* p01 = (const float4*)(srcb + (size_t)(yc1*Ww + xc0)*Cc);
    const float4* p11 = (const float4*)(srcb + (size_t)(yc1*Ww + xc1)*Cc);
    float acc_s = 0.f, acc_c = 0.f;
#pragma unroll
    for (int i = 0; i < 8; i++) {
      float4 a = p00[i], b = p10[i], c = p01[i], e = p11[i];
      float sx = w00*a.x + w10*b.x + w01*c.x + w11*e.x;
      float sy = w00*a.y + w10*b.y + w01*c.y + w11*e.y;
      float sz = w00*a.z + w10*b.z + w01*c.z + w11*e.z;
      float sw = w00*a.w + w10*b.w + w01*c.w + w11*e.w;
      acc_s += sx*rc[i].x + sy*rc[i].y + sz*rc[i].z + sw*rc[i].w;
      acc_c += sx*cw[i].x + sy*cw[i].y + sz*cw[i].z + sw*cw[i].w;
    }
    sim  [((size_t)v*Dd + d)*HW + n] = acc_s;
    cpart[((size_t)v*Dd + d)*HW + n] = acc_c;
  }
}

// ---------------------------------------------------------------------------
// K3: softmax entropy over D, emit f16 CNN inputs.
// ---------------------------------------------------------------------------
__global__ void entropy_kernel(const float* __restrict__ sim, const float* __restrict__ ref_nc,
                               _Float16* __restrict__ ent, _Float16* __restrict__ refnc_h) {
  int n = blockIdx.x * 256 + threadIdx.x;
  int v = blockIdx.y;
  const float* s = sim + (size_t)v*Dd*HW + n;
  float m = -3.0e38f;
  for (int d = 0; d < Dd; d++) m = fmaxf(m, s[(size_t)d*HW]);
  float se = 0.f, ses = 0.f;
  for (int d = 0; d < Dd; d++) {
    float sv = s[(size_t)d*HW];
    float e = __expf(sv - m);
    se += e; ses += e*sv;
  }
  float L = m + __logf(se);
  ent[(size_t)v*HW + n] = (_Float16)(L - ses/se);
  if (v == 0) refnc_h[n] = (_Float16)ref_nc[n];
}

// ---------------------------------------------------------------------------
// K4: pack conv weights into the WMMA f16 A-matrix lane layout
// (ISA 7.12.2: lane<16 rows M=L holding K={e:e<8, e+8:e>=8}; lane>=16 rows
// M=L-16 holding K={e+8:e<8, e+16:e>=8} per 32-K chunk). K = tap*Cin + cin.
// Sections: L1 @0 (512 halfs), L2 @512 (2560), L3 @3072 (2560).
// ---------------------------------------------------------------------------
__global__ void packw_kernel(const float* __restrict__ w1, const float* __restrict__ w2,
                             const float* __restrict__ w3, _Float16* __restrict__ wp) {
  int idx = blockIdx.x * 256 + threadIdx.x;
  if (idx >= 5632) return;
  const float* w; int Cin, Ktot, q, rem;
  if (idx < 512)       { w = w1; Cin = 2;  Ktot = 18;  rem = idx;        q = 0; }
  else if (idx < 3072) { w = w2; Cin = 16; Ktot = 144; rem = idx - 512;  q = rem/512; rem %= 512; }
  else                 { w = w3; Cin = 16; Ktot = 144; rem = idx - 3072; q = rem/512; rem %= 512; }
  int lane = rem / 16, e = rem % 16;
  int Kl = (lane < 16) ? (e < 8 ? e : e + 8) : (e < 8 ? e + 8 : e + 16);
  int K  = q*32 + Kl;
  int M  = lane & 15;
  _Float16 val = (_Float16)0.f;
  if (K < Ktot) {
    int tap = K / Cin, cin = K % Cin;
    val = (_Float16)w[(M*Cin + cin)*9 + tap];
  }
  wp[idx] = val;
}

// ---------------------------------------------------------------------------
// K5: conv1 (Cin=2 -> 16, 3x3 pad 1) + scale/bias + relu, WMMA f32_16x16x32_f16.
// One wave per block computes a 16-outch x 16-pixel tile.
// B layout (matches packed A): lane<16 -> col L, K={e:e<8, e+8:e>=8};
// lane>=16 -> col L-16, K={e+8:e<8, e+16:e>=8}.
// ---------------------------------------------------------------------------
__global__ void conv1_kernel(const _Float16* __restrict__ ent, const _Float16* __restrict__ refnc,
                             const _Float16* __restrict__ wp,
                             const float* __restrict__ sc, const float* __restrict__ tb,
                             _Float16* __restrict__ xout) {
  __shared__ _Float16 tile[2*3*18];
  int v = blockIdx.y;
  int t = blockIdx.x;
  int yrow = t / (Ww/16);
  int x0p  = (t % (Ww/16)) * 16;
  int lane = threadIdx.x;
  const _Float16* ch0 = ent + (size_t)v*HW;
  for (int idx = lane; idx < 2*3*18; idx += 32) {
    int c = idx / 54, r = idx % 54, ry = r / 18, rx2 = r % 18;
    int gy = yrow - 1 + ry, gx = x0p - 1 + rx2;
    _Float16 val = (_Float16)0.f;
    if (gy >= 0 && gy < Hh && gx >= 0 && gx < Ww)
      val = (c == 0) ? ch0[gy*Ww + gx] : refnc[gy*Ww + gx];
    tile[idx] = val;
  }
  __syncthreads();
  v16h a = *(const v16h*)(wp + lane*16);
  v16h b;
#pragma unroll
  for (int e = 0; e < 16; e++) {
    int K = (lane < 16) ? (e < 8 ? e : e + 8) : (e < 8 ? e + 8 : e + 16);
    _Float16 val = (_Float16)0.f;
    if (K < 18) {
      int tap = K >> 1, cin = K & 1;
      val = tile[cin*54 + (tap/3)*18 + (tap%3) + (lane & 15)];
    }
    b[e] = val;
  }
  v8f c = {};
  c = __builtin_amdgcn_wmma_f32_16x16x32_f16(false, a, false, b, (short)0, c, false, false);
  int col = lane & 15, chb = (lane < 16) ? 0 : 8;
#pragma unroll
  for (int r = 0; r < 8; r++) {
    int oc = chb + r;
    float y = fmaxf(c[r]*sc[oc] + tb[oc], 0.f);
    xout[((size_t)v*16 + oc)*HW + yrow*Ww + x0p + col] = (_Float16)y;
  }
}

// ---------------------------------------------------------------------------
// K6/K7: conv 16->16 3x3 pad1 (+relu). K=144 -> 5 WMMA chunks of 32.
// fuse=1: fold the 1x1 conv (w4) + bias + sigmoid into the epilogue via a
// cross-half __shfl_xor(16) reduction and emit vis directly.
// ---------------------------------------------------------------------------
__global__ void conv16_kernel(const _Float16* __restrict__ xin, const _Float16* __restrict__ wp,
                              int wpOff,
                              const float* __restrict__ sc, const float* __restrict__ tb,
                              _Float16* __restrict__ xout,
                              const float* __restrict__ w4, const float* __restrict__ b4,
                              float* __restrict__ vis, int fuse) {
  __shared__ _Float16 tile[16*54];
  int v = blockIdx.y;
  int t = blockIdx.x;
  int yrow = t / (Ww/16);
  int x0p  = (t % (Ww/16)) * 16;
  int lane = threadIdx.x;
  const _Float16* in = xin + (size_t)v*16*HW;
  for (int idx = lane; idx < 16*54; idx += 32) {
    int c = idx / 54, r = idx % 54, ry = r / 18, rx2 = r % 18;
    int gy = yrow - 1 + ry, gx = x0p - 1 + rx2;
    _Float16 val = (_Float16)0.f;
    if (gy >= 0 && gy < Hh && gx >= 0 && gx < Ww)
      val = in[(size_t)c*HW + gy*Ww + gx];
    tile[idx] = val;
  }
  __syncthreads();
  v8f c = {};
#pragma unroll
  for (int q = 0; q < 5; q++) {
    v16h a = *(const v16h*)(wp + wpOff + (q*32 + lane)*16);
    v16h b;
#pragma unroll
    for (int e = 0; e < 16; e++) {
      int K = q*32 + ((lane < 16) ? (e < 8 ? e : e + 8) : (e < 8 ? e + 8 : e + 16));
      _Float16 val = (_Float16)0.f;
      if (K < 144) {
        int tap = K >> 4, cin = K & 15;
        val = tile[cin*54 + (tap/3)*18 + (tap%3) + (lane & 15)];
      }
      b[e] = val;
    }
    c = __builtin_amdgcn_wmma_f32_16x16x32_f16(false, a, false, b, (short)0, c, false, false);
  }
  int col = lane & 15, chb = (lane < 16) ? 0 : 8;
  if (!fuse) {
#pragma unroll
    for (int r = 0; r < 8; r++) {
      int oc = chb + r;
      float y = fmaxf(c[r]*sc[oc] + tb[oc], 0.f);
      xout[((size_t)v*16 + oc)*HW + yrow*Ww + x0p + col] = (_Float16)y;
    }
  } else {
    float part = 0.f;
#pragma unroll
    for (int r = 0; r < 8; r++) {
      int oc = chb + r;
      float y = fmaxf(c[r]*sc[oc] + tb[oc], 0.f);
      part += w4[oc]*y;
    }
    float tot = part + __shfl_xor(part, 16, 32) + b4[0];
    float vv = 1.f / (1.f + __expf(-tot));
    if (lane < 16)
      vis[(size_t)v*HW + yrow*Ww + x0p + col] = vv;
  }
}

// ---------------------------------------------------------------------------
// K8: per-pixel finalize: cost softmax over D (held in VGPRs), prob, depth,
// 4-window confidence, nc_mean.
// ---------------------------------------------------------------------------
__global__ void final_kernel(const float* __restrict__ cpart, const float* __restrict__ vis,
                             const float* __restrict__ dvals,
                             const float* __restrict__ ref_nc_sum, const float* __restrict__ src_nc_sums,
                             const float* __restrict__ reg_b, float* __restrict__ out) {
  int n = blockIdx.x * 256 + threadIdx.x;
  float v0 = vis[n], v1 = vis[HW + n], v2 = vis[2*HW + n];
  float inv = 1.f / (v0 + v1 + v2 + 1e-6f);
  float rb = reg_b[0];
  float p[Dd];
  float m = -3.0e38f;
#pragma unroll
  for (int d = 0; d < Dd; d++) {
    float cc = (v0*cpart[(size_t)(0*Dd + d)*HW + n]
              + v1*cpart[(size_t)(1*Dd + d)*HW + n]
              + v2*cpart[(size_t)(2*Dd + d)*HW + n]) * inv + rb;
    p[d] = cc;
    m = fmaxf(m, cc);
  }
  float se = 0.f;
#pragma unroll
  for (int d = 0; d < Dd; d++) { p[d] = __expf(p[d] - m); se += p[d]; }
  float ise = 1.f / se;
  float depth = 0.f, widx = 0.f;
#pragma unroll
  for (int d = 0; d < Dd; d++) {
    p[d] *= ise;
    out[(size_t)2*HW + (size_t)d*HW + n] = p[d];
    depth += p[d] * dvals[(size_t)d*HW + n];
    widx  += p[d] * (float)d;
  }
  int idx = (int)widx;
  idx = idx < 0 ? 0 : (idx > Dd-1 ? Dd-1 : idx);
  float conf = 0.f;
#pragma unroll
  for (int d = 0; d < Dd; d++)
    if (d >= idx - 1 && d <= idx + 2) conf += p[d];
  out[n]      = depth;
  out[HW + n] = conf;
  float ssum = src_nc_sums[n] + src_nc_sums[HW + n] + src_nc_sums[2*HW + n];
  out[(size_t)(2 + Dd)*HW + n] = ref_nc_sum[n]*0.5f + ssum*(1.f/6.f);
}

// ---------------------------------------------------------------------------
static inline size_t alignup(size_t x) { return (x + 255) & ~(size_t)255; }

extern "C" void kernel_launch(void* const* d_in, const int* in_sizes, int n_in,
                              void* d_out, int out_size, void* d_ws, size_t ws_size,
                              hipStream_t stream) {
  const float* ref_fea     = (const float*)d_in[0];
  const float* src_feas    = (const float*)d_in[1];
  const float* ref_nc      = (const float*)d_in[2];
  const float* ref_nc_sum  = (const float*)d_in[3];
  const float* src_nc_sums = (const float*)d_in[4];
  const float* proj_mats   = (const float*)d_in[5];
  const float* depth_vals  = (const float*)d_in[6];
  const float* vis_w1 = (const float*)d_in[7];
  const float* vis_s1 = (const float*)d_in[8];
  const float* vis_t1 = (const float*)d_in[9];
  const float* vis_w2 = (const float*)d_in[10];
  const float* vis_s2 = (const float*)d_in[11];
  const float* vis_t2 = (const float*)d_in[12];
  const float* vis_w3 = (const float*)d_in[13];
  const float* vis_s3 = (const float*)d_in[14];
  const float* vis_t3 = (const float*)d_in[15];
  const float* vis_w4 = (const float*)d_in[16];
  const float* vis_b4 = (const float*)d_in[17];
  const float* reg_w  = (const float*)d_in[18];
  const float* reg_b  = (const float*)d_in[19];
  float* out = (float*)d_out;

  char* ws = (char*)d_ws;
  size_t o = 0;
  float* projw = (float*)(ws + o);      o = alignup(o + 48*sizeof(float));
  float* hwc   = (float*)(ws + o);      o = alignup(o + (size_t)4*HW*Cc*sizeof(float));
  float* sim   = (float*)(ws + o);      o = alignup(o + (size_t)NVIEW*Dd*HW*sizeof(float));
  float* cpart = (float*)(ws + o);      o = alignup(o + (size_t)NVIEW*Dd*HW*sizeof(float));
  float* vis   = (float*)(ws + o);      o = alignup(o + (size_t)NVIEW*HW*sizeof(float));
  _Float16* ent   = (_Float16*)(ws + o); o = alignup(o + (size_t)NVIEW*HW*sizeof(_Float16));
  _Float16* refnc = (_Float16*)(ws + o); o = alignup(o + (size_t)HW*sizeof(_Float16));
  _Float16* wp    = (_Float16*)(ws + o); o = alignup(o + (size_t)5632*sizeof(_Float16));
  _Float16* x1    = (_Float16*)(ws + o); o = alignup(o + (size_t)NVIEW*16*HW*sizeof(_Float16));
  _Float16* x2    = (_Float16*)(ws + o); o = alignup(o + (size_t)NVIEW*16*HW*sizeof(_Float16));
  (void)o; (void)ws_size; (void)in_sizes; (void)n_in; (void)out_size;

  proj_kernel<<<1, 32, 0, stream>>>(proj_mats, projw);
  chw2hwc_kernel<<<dim3(HW/64, 4), 256, 0, stream>>>(ref_fea, src_feas, hwc);
  packw_kernel<<<22, 256, 0, stream>>>(vis_w1, vis_w2, vis_w3, wp);
  warp_kernel<<<dim3(HW/256, 8, NVIEW), 256, 0, stream>>>(projw, hwc, reg_w, depth_vals, sim, cpart);
  entropy_kernel<<<dim3(HW/256, NVIEW), 256, 0, stream>>>(sim, ref_nc, ent, refnc);
  conv1_kernel<<<dim3((HW/16), NVIEW), 32, 0, stream>>>(ent, refnc, wp, vis_s1, vis_t1, x1);
  conv16_kernel<<<dim3((HW/16), NVIEW), 32, 0, stream>>>(x1, wp, 512,  vis_s2, vis_t2, x2,
                                                         vis_w4, vis_b4, vis, 0);
  conv16_kernel<<<dim3((HW/16), NVIEW), 32, 0, stream>>>(x2, wp, 3072, vis_s3, vis_t3, x1,
                                                         vis_w4, vis_b4, vis, 1);
  final_kernel<<<HW/256, 256, 0, stream>>>(cpart, vis, depth_vals, ref_nc_sum, src_nc_sums,
                                           reg_b, out);
}
